// DocREModel_15633680957873
// MI455X (gfx1250) — matrix operand order repaired and checked by reading, not compiled, not run on hardware
//
#include <hip/hip_runtime.h>

// ---------------- problem constants (from reference) ----------------
constexpr int N_ = 4, C_ = 512, D_ = 768, H_ = 12, E_ = 20, M_ = 4;
constexpr int P_ = E_ * (E_ - 1);          // 380
constexpr int EMB_ = 768, BLK_ = 64, L_ = 97;
constexpr float NEGV_ = -1e30f;

typedef _Float16 h16;
typedef __attribute__((ext_vector_type(16))) _Float16 v16h;
typedef __attribute__((ext_vector_type(8)))  _Float16 v8h;
typedef __attribute__((ext_vector_type(8)))  float    v8f;

static __device__ __forceinline__ v16h join8(v8h lo, v8h hi) {
    return __builtin_shufflevector(lo, hi, 0, 1, 2, 3, 4, 5, 6, 7,
                                           8, 9, 10, 11, 12, 13, 14, 15);
}
static __device__ __forceinline__ v16h ldfrag(const h16* p0, const h16* p1) {
    return join8(*(const v8h*)p0, *(const v8h*)p1);
}

// ---------------- small helper kernels ----------------

// transpose + f32->f16: src is K x Nn row-major, dst is Nn x K row-major; batched via blockIdx.z
__global__ void k_f2h_t(const float* __restrict__ src, h16* __restrict__ dst, int K, int Nn) {
    int i = blockIdx.x * blockDim.x + threadIdx.x;
    if (i >= K * Nn) return;
    size_t base = (size_t)blockIdx.z * K * Nn;
    int k = i / Nn, n = i % Nn;
    dst[base + (size_t)n * K + k] = (h16)src[base + i];
}

// gather e_emb = seq[pos], e_att = mean_h att[:,pos,:] (+pad col); one block per (n,e,m)
__global__ void k_gather(const float* __restrict__ seq, const float* __restrict__ att,
                         const int* __restrict__ ms,
                         float* __restrict__ e_emb, float* __restrict__ e_att) {
    int nem = blockIdx.x;                 // n*E*M + e*M + m
    int n   = nem / (E_ * M_);
    int pos = ms[nem] + 1;
    const float* srow = seq + ((size_t)n * C_ + pos) * D_;
    float* erow = e_emb + (size_t)nem * D_;
    for (int d = threadIdx.x; d < D_; d += blockDim.x) erow[d] = srow[d];
    float* arow = e_att + (size_t)nem * (C_ + 1);
    const float* abase = att + ((size_t)n * H_ * C_ + pos) * C_;
    for (int c = threadIdx.x; c < C_; c += blockDim.x) {
        float s = 0.f;
        for (int h = 0; h < H_; ++h) s += abase[(size_t)h * C_ * C_ + c];
        arow[c] = s * (1.0f / H_);
    }
    if (threadIdx.x == 0) arow[C_] = 0.f;
}

// glob = logsumexp over M mentions, fp32 + f16 copy for the WMMA GEMM
__global__ void k_glob(const float* __restrict__ e_emb, float* __restrict__ glob_h32,
                       h16* __restrict__ glob_h) {
    int i = blockIdx.x * blockDim.x + threadIdx.x;   // over N*E*D
    if (i >= N_ * E_ * D_) return;
    int d = i % D_;
    int ne = i / D_;
    const float* base = e_emb + (size_t)ne * M_ * D_ + d;
    float mx = -3.0e38f;
    for (int m = 0; m < M_; ++m) { float v = base[m * D_]; mx = fmaxf(mx, v); }
    float s = 0.f;
    for (int m = 0; m < M_; ++m) s += __expf(base[m * D_] - mx);
    float v = mx + __logf(s);
    glob_h32[i] = v;
    glob_h[i] = (h16)v;
}

// pair attention weights (P x M), normalized; one thread per (n,p)
__global__ void k_pairw(const float* __restrict__ e_att, const int* __restrict__ ms,
                        const int* __restrict__ hts,
                        float* __restrict__ p_hatt, float* __restrict__ p_tatt) {
    int np = blockIdx.x * blockDim.x + threadIdx.x;
    if (np >= N_ * P_) return;
    int n = np / P_;
    int h = hts[(size_t)np * 2 + 0];
    int t = hts[(size_t)np * 2 + 1];
    const int* pos_n = ms + n * E_ * M_;
    const float* eatt_n = e_att + (size_t)n * E_ * M_ * (C_ + 1);
    float ph[M_], pt[M_];
    float sh = 0.f, st = 0.f;
    for (int mm = 0; mm < M_; ++mm) {
        int ptp = pos_n[t * M_ + mm] + 1;
        int php = pos_n[h * M_ + mm] + 1;
        float a = 0.f, b = 0.f;
        for (int m = 0; m < M_; ++m) {
            a += eatt_n[(size_t)(h * M_ + m) * (C_ + 1) + ptp];
            b += eatt_n[(size_t)(t * M_ + m) * (C_ + 1) + php];
        }
        a *= (1.f / M_); b *= (1.f / M_);
        ph[mm] = a; pt[mm] = b; sh += a; st += b;
    }
    for (int mm = 0; mm < M_; ++mm) {
        p_hatt[(size_t)np * M_ + mm] = ph[mm] / (sh + 1e-5f);
        p_tatt[(size_t)np * M_ + mm] = pt[mm] / (st + 1e-5f);
    }
}

// new_hs/new_ts (fp32) and packed q = [new_hs|new_ts] in f16; one block per (n,p)
__global__ void k_newht(const float* __restrict__ e_emb, const float* __restrict__ p_hatt,
                        const float* __restrict__ p_tatt, const int* __restrict__ hts,
                        float* __restrict__ new_hs, float* __restrict__ new_ts,
                        h16* __restrict__ q_h) {
    int np = blockIdx.x;
    int n = np / P_;
    int h = hts[(size_t)np * 2 + 0];
    int t = hts[(size_t)np * 2 + 1];
    const float* eh = e_emb + ((size_t)n * E_ + h) * M_ * D_;
    const float* et = e_emb + ((size_t)n * E_ + t) * M_ * D_;
    float wt[M_], wh[M_];
    for (int m = 0; m < M_; ++m) { wt[m] = p_tatt[(size_t)np * M_ + m]; wh[m] = p_hatt[(size_t)np * M_ + m]; }
    for (int d = threadIdx.x; d < D_; d += blockDim.x) {
        float a = 0.f, b = 0.f;
        for (int m = 0; m < M_; ++m) { a += wt[m] * eh[m * D_ + d]; b += wh[m] * et[m * D_ + d]; }
        new_hs[(size_t)np * D_ + d] = a;
        new_ts[(size_t)np * D_ + d] = b;
        q_h[(size_t)np * 2 * D_ + d]      = (h16)a;
        q_h[(size_t)np * 2 * D_ + D_ + d] = (h16)b;
    }
}

// pa = normalize(nh_att * nt_att), written as f16 (first 512 cols for rs GEMM)
__global__ void k_pa(const float* __restrict__ e_att, const float* __restrict__ p_hatt,
                     const float* __restrict__ p_tatt, const int* __restrict__ hts,
                     float* __restrict__ pa_tmp, h16* __restrict__ pa_h) {
    __shared__ float red[256];
    int np = blockIdx.x;
    int n = np / P_;
    int h = hts[(size_t)np * 2 + 0];
    int t = hts[(size_t)np * 2 + 1];
    const float* ah = e_att + ((size_t)n * E_ + h) * M_ * (C_ + 1);
    const float* at = e_att + ((size_t)n * E_ + t) * M_ * (C_ + 1);
    float wt[M_], wh[M_];
    for (int m = 0; m < M_; ++m) { wt[m] = p_tatt[(size_t)np * M_ + m]; wh[m] = p_hatt[(size_t)np * M_ + m]; }
    float lsum = 0.f;
    for (int c = threadIdx.x; c < C_; c += blockDim.x) {   // padded col C_ contributes 0
        float a = 0.f, b = 0.f;
        for (int m = 0; m < M_; ++m) {
            a += wt[m] * ah[m * (C_ + 1) + c];
            b += wh[m] * at[m * (C_ + 1) + c];
        }
        float v = a * b;
        pa_tmp[(size_t)np * C_ + c] = v;
        lsum += v;
    }
    red[threadIdx.x] = lsum;
    __syncthreads();
    for (int s = 128; s > 0; s >>= 1) {
        if (threadIdx.x < s) red[threadIdx.x] += red[threadIdx.x + s];
        __syncthreads();
    }
    float inv = 1.f / (red[0] + 1e-5f);
    for (int c = threadIdx.x; c < C_; c += blockDim.x)
        pa_h[(size_t)np * C_ + c] = (h16)(pa_tmp[(size_t)np * C_ + c] * inv);
}

// edge = relu(A[e1] + B[e2] + bm)
__global__ void k_edge(const float* __restrict__ Am, const float* __restrict__ Bm,
                       const float* __restrict__ bm, float* __restrict__ edge) {
    int i = blockIdx.x * blockDim.x + threadIdx.x;
    if (i >= N_ * E_ * E_ * D_) return;
    int d = i % D_;
    int r = i / D_;
    int e2 = r % E_; r /= E_;
    int e1 = r % E_; int n = r / E_;
    float v = Am[((size_t)n * E_ + e1) * D_ + d] + Bm[((size_t)n * E_ + e2) * D_ + d] + bm[d];
    edge[i] = fmaxf(v, 0.f);
}

// score[np,v] = sum_k <q_k, E_k[p,v,:]> + batt ; one thread per (np,v)
__global__ void k_score(const float* __restrict__ qW, const float* __restrict__ edge,
                        const int* __restrict__ hts, const float* __restrict__ batt,
                        float* __restrict__ score) {
    int idx = blockIdx.x * blockDim.x + threadIdx.x;
    if (idx >= N_ * P_ * E_) return;
    int v = idx % E_;
    int np = idx / E_;
    int n = np / P_;
    int h = hts[(size_t)np * 2 + 0];
    int t = hts[(size_t)np * 2 + 1];
    const float* q = qW + (size_t)np * 4 * D_;
    const float* eb = edge + (size_t)n * E_ * E_ * D_;
    const float* e1 = eb + ((size_t)h * E_ + v) * D_;
    const float* e2 = eb + ((size_t)v * E_ + t) * D_;
    const float* e3 = eb + ((size_t)t * E_ + v) * D_;
    const float* e4 = eb + ((size_t)v * E_ + h) * D_;
    float s = 0.f;
    for (int d = 0; d < D_; ++d)
        s += q[d] * e1[d] + q[D_ + d] * e2[d] + q[2 * D_ + d] * e3[d] + q[3 * D_ + d] * e4[d];
    score[idx] = s + batt[0];
}

// masked softmax over E; one thread per (n,p)
__global__ void k_soft(const float* __restrict__ score, const int* __restrict__ hts,
                       float* __restrict__ aw) {
    int np = blockIdx.x * blockDim.x + threadIdx.x;
    if (np >= N_ * P_) return;
    int h = hts[(size_t)np * 2 + 0];
    int t = hts[(size_t)np * 2 + 1];
    float sc[E_];
    float mx = -3.0e38f;
    for (int v = 0; v < E_; ++v) {
        float s = (v == h || v == t) ? NEGV_ : score[(size_t)np * E_ + v];
        sc[v] = s;
        mx = fmaxf(mx, s);
    }
    float sum = 0.f;
    for (int v = 0; v < E_; ++v) { sc[v] = __expf(sc[v] - mx); sum += sc[v]; }
    for (int v = 0; v < E_; ++v) aw[(size_t)np * E_ + v] = sc[v] / sum;
}

// path[np, k*D+d] = sum_v aw[v]*E_k[v,d]  -> written directly as f16 GEMM input
__global__ void k_path(const float* __restrict__ aw, const float* __restrict__ edge,
                       const int* __restrict__ hts, h16* __restrict__ path_h) {
    int np = blockIdx.x;
    int n = np / P_;
    int h = hts[(size_t)np * 2 + 0];
    int t = hts[(size_t)np * 2 + 1];
    const float* eb = edge + (size_t)n * E_ * E_ * D_;
    float w[E_];
    for (int v = 0; v < E_; ++v) w[v] = aw[(size_t)np * E_ + v];
    h16* pr = path_h + (size_t)np * 4 * D_;
    for (int d = threadIdx.x; d < D_; d += blockDim.x) {
        float s1 = 0.f, s2 = 0.f, s3 = 0.f, s4 = 0.f;
        for (int v = 0; v < E_; ++v) {
            s1 += w[v] * eb[((size_t)h * E_ + v) * D_ + d];
            s2 += w[v] * eb[((size_t)v * E_ + t) * D_ + d];
            s3 += w[v] * eb[((size_t)t * E_ + v) * D_ + d];
            s4 += w[v] * eb[((size_t)v * E_ + h) * D_ + d];
        }
        pr[d] = (h16)s1; pr[D_ + d] = (h16)s2; pr[2 * D_ + d] = (h16)s3; pr[3 * D_ + d] = (h16)s4;
    }
}

// build f16 concat inputs [nh|rs|path2] and [nt|rs|path2]
__global__ void k_cat(const float* __restrict__ nh, const float* __restrict__ nt,
                      const float* __restrict__ rs, const float* __restrict__ p2,
                      h16* __restrict__ ch, h16* __restrict__ ct) {
    int i = blockIdx.x * blockDim.x + threadIdx.x;     // N*P*D
    if (i >= N_ * P_ * D_) return;
    int d = i % D_;
    int np = i / D_;
    size_t base = (size_t)np * 3 * D_;
    h16 r = (h16)rs[i], p = (h16)p2[i];
    ch[base + d] = (h16)nh[i]; ch[base + D_ + d] = r; ch[base + 2 * D_ + d] = p;
    ct[base + d] = (h16)nt[i]; ct[base + D_ + d] = r; ct[base + 2 * D_ + d] = p;
}

// ---------------- WMMA GEMM v2 ----------------
// A: Mr x K row-major f16 (lda, mult of 8). Bt: Nc x K row-major f16 (TRANSPOSED B).
// One wave per block; 32x32 C tile (2 A frags x 2 B frags -> 4 wmma per K-step).
// All fragment loads are contiguous 16B (global_load_b128).
__global__ void k_gemm(const h16* __restrict__ A, const h16* __restrict__ Bt,
                       const float* __restrict__ bias, float* __restrict__ Cm,
                       int Mr, int Nc, int K, int lda, int ldc,
                       long long sA, long long sB, long long sC, int relu) {
    int lane = threadIdx.x;
    int tn = blockIdx.x * 32, tm = blockIdx.y * 32;
    A  += (size_t)blockIdx.z * sA;
    Bt += (size_t)blockIdx.z * sB;
    Cm += (size_t)blockIdx.z * sC;
    int ml = lane & 15, kh = lane >> 4;
    int r0 = tm + ml;      r0 = r0 < Mr ? r0 : Mr - 1;     // clamp: garbage rows/cols never stored
    int r1 = tm + 16 + ml; r1 = r1 < Mr ? r1 : Mr - 1;
    int c0 = tn + ml;      int c0c = c0 < Nc ? c0 : Nc - 1;
    int c1 = tn + 16 + ml; int c1c = c1 < Nc ? c1 : Nc - 1;
    const h16* a0p = A  + (size_t)r0 * lda + 8 * kh;
    const h16* a1p = A  + (size_t)r1 * lda + 8 * kh;
    const h16* b0p = Bt + (size_t)c0c * K + 8 * kh;
    const h16* b1p = Bt + (size_t)c1c * K + 8 * kh;
    v8f acc00 = {}, acc01 = {}, acc10 = {}, acc11 = {};
    for (int k0 = 0; k0 < K; k0 += 32) {
        v16h a0 = ldfrag(a0p + k0, a0p + k0 + 16);
        v16h a1 = ldfrag(a1p + k0, a1p + k0 + 16);
        v16h b0 = ldfrag(b0p + k0, b0p + k0 + 16);
        v16h b1 = ldfrag(b1p + k0, b1p + k0 + 16);
        __builtin_prefetch((const void*)(a0p + k0 + 256), 0, 0);   // global_prefetch_b8
        __builtin_prefetch((const void*)(b0p + k0 + 256), 0, 0);
        acc00 = __builtin_amdgcn_wmma_f32_16x16x32_f16(false, a0, false, b0, (short)0, acc00, false, false);
        acc01 = __builtin_amdgcn_wmma_f32_16x16x32_f16(false, a0, false, b1, (short)0, acc01, false, false);
        acc10 = __builtin_amdgcn_wmma_f32_16x16x32_f16(false, a1, false, b0, (short)0, acc10, false, false);
        acc11 = __builtin_amdgcn_wmma_f32_16x16x32_f16(false, a1, false, b1, (short)0, acc11, false, false);
    }
#pragma unroll
    for (int r = 0; r < 8; ++r) {
        int mA = tm + r + 8 * kh;
        int mB = tm + 16 + r + 8 * kh;
        if (mA < Mr) {
            if (c0 < Nc) { float v = acc00[r] + (bias ? bias[c0] : 0.f); if (relu) v = fmaxf(v, 0.f); Cm[(size_t)mA * ldc + c0] = v; }
            if (c1 < Nc) { float v = acc01[r] + (bias ? bias[c1] : 0.f); if (relu) v = fmaxf(v, 0.f); Cm[(size_t)mA * ldc + c1] = v; }
        }
        if (mB < Mr) {
            if (c0 < Nc) { float v = acc10[r] + (bias ? bias[c0] : 0.f); if (relu) v = fmaxf(v, 0.f); Cm[(size_t)mB * ldc + c0] = v; }
            if (c1 < Nc) { float v = acc11[r] + (bias ? bias[c1] : 0.f); if (relu) v = fmaxf(v, 0.f); Cm[(size_t)mB * ldc + c1] = v; }
        }
    }
}

// ---------------- fused bilinear classifier GEMM ----------------
// out = bl @ Wbil + bbil with bl[n, k*4096+i*64+j] = hs[n,k*64+i]*ts[n,k*64+j] generated on
// the fly (1520x49152 never materialized). Bt = Wbil transposed (97 x 49152) f16.
// B tiles are staged LDS-side with a double-buffered global_load_async_to_lds_b128 pipeline
// (ASYNCcnt) so the copy of step k+1 overlaps the 4 WMMAs of step k; fragments come back
// via ds_load_b128.
__global__ void k_gemm_bil(const float* __restrict__ hs, const float* __restrict__ ts,
                           const h16* __restrict__ Bt, const float* __restrict__ bias,
                           float* __restrict__ Cm, int Mr, int Nc) {
    constexpr int K = EMB_ * BLK_;         // 49152
    __shared__ alignas(16) h16 smB[2][1024];   // 2 x (32 cols x 32 k) f16, col-major [col][k]
    int lane = threadIdx.x;
    int tn = blockIdx.x * 32, tm = blockIdx.y * 32;
    int ml = lane & 15, kh = lane >> 4;
    int r0 = tm + ml;      r0 = r0 < Mr ? r0 : Mr - 1;
    int r1 = tm + 16 + ml; r1 = r1 < Mr ? r1 : Mr - 1;
    const float* h0 = hs + (size_t)r0 * D_;
    const float* t0 = ts + (size_t)r0 * D_;
    const float* h1 = hs + (size_t)r1 * D_;
    const float* t1 = ts + (size_t)r1 * D_;

    auto issueB = [&](int k0, int bufsel) {
#pragma unroll
        for (int q = 0; q < 4; ++q) {                 // 4 x 512B = 2KB per buffer
            int chunk = q * 32 + lane;                // 128 x 16B chunks
            int cl  = chunk >> 2;                     // col_local 0..31
            int sub = chunk & 3;                      // 16B piece within the 64B column
            int col = tn + cl; col = col < Nc ? col : Nc - 1;
            const h16* src = Bt + (size_t)col * K + k0 + sub * 8;
            unsigned loff = (unsigned)(size_t)&smB[bufsel][cl * 32 + sub * 8];
            unsigned long long ga = (unsigned long long)(size_t)src;
            asm volatile("global_load_async_to_lds_b128 %0, %1, off"
                         :: "v"(loff), "v"(ga) : "memory");
        }
    };

    v8f acc00 = {}, acc01 = {}, acc10 = {}, acc11 = {};
    issueB(0, 0);
    int cur = 0;
    for (int k0 = 0; k0 < K; k0 += 32) {
        if (k0 + 32 < K) {
            issueB(k0 + 32, cur ^ 1);
            asm volatile("s_wait_asynccnt 0x4" ::: "memory");   // current buffer landed (in-order)
        } else {
            asm volatile("s_wait_asynccnt 0x0" ::: "memory");
        }
        v16h a0, a1;
#pragma unroll
        for (int j = 0; j < 16; ++j) {
            int kg = k0 + j + 16 * kh;
            int blk = kg >> 12;
            int ii = (kg >> 6) & 63;
            int jj = kg & 63;
            a0[j] = (h16)(h0[blk * 64 + ii] * t0[blk * 64 + jj]);
            a1[j] = (h16)(h1[blk * 64 + ii] * t1[blk * 64 + jj]);
        }
        const h16* sb = &smB[cur][0];
        v16h b0 = ldfrag(sb + ml * 32 + 16 * kh,        sb + ml * 32 + 16 * kh + 8);
        v16h b1 = ldfrag(sb + (16 + ml) * 32 + 16 * kh, sb + (16 + ml) * 32 + 16 * kh + 8);
        acc00 = __builtin_amdgcn_wmma_f32_16x16x32_f16(false, a0, false, b0, (short)0, acc00, false, false);
        acc01 = __builtin_amdgcn_wmma_f32_16x16x32_f16(false, a0, false, b1, (short)0, acc01, false, false);
        acc10 = __builtin_amdgcn_wmma_f32_16x16x32_f16(false, a1, false, b0, (short)0, acc10, false, false);
        acc11 = __builtin_amdgcn_wmma_f32_16x16x32_f16(false, a1, false, b1, (short)0, acc11, false, false);
        cur ^= 1;
    }
#pragma unroll
    for (int r = 0; r < 8; ++r) {
        int mA = tm + r + 8 * kh;
        int mB = tm + 16 + r + 8 * kh;
        int c0 = tn + ml, c1 = tn + 16 + ml;
        if (mA < Mr) {
            if (c0 < Nc) Cm[(size_t)mA * L_ + c0] = acc00[r] + bias[c0];
            if (c1 < Nc) Cm[(size_t)mA * L_ + c1] = acc01[r] + bias[c1];
        }
        if (mB < Mr) {
            if (c0 < Nc) Cm[(size_t)mB * L_ + c0] = acc10[r] + bias[c0];
            if (c1 < Nc) Cm[(size_t)mB * L_ + c1] = acc11[r] + bias[c1];
        }
    }
}

// ---------------- host orchestration ----------------
extern "C" void kernel_launch(void* const* d_in, const int* in_sizes, int n_in,
                              void* d_out, int out_size, void* d_ws, size_t ws_size,
                              hipStream_t stream) {
    const float* seq   = (const float*)d_in[0];
    const float* att   = (const float*)d_in[1];
    const int*   ms    = (const int*)d_in[2];
    const int*   hts   = (const int*)d_in[3];
    const float* Wm1   = (const float*)d_in[4];
    const float* Wm2   = (const float*)d_in[5];
    const float* bm    = (const float*)d_in[6];
    const float* Watt  = (const float*)d_in[7];
    const float* batt  = (const float*)d_in[8];
    const float* Wpath = (const float*)d_in[9];
    const float* bpath = (const float*)d_in[10];
    const float* Whead = (const float*)d_in[11];
    const float* bhead = (const float*)d_in[12];
    const float* Wtail = (const float*)d_in[13];
    const float* btail = (const float*)d_in[14];
    const float* Wbil  = (const float*)d_in[15];
    const float* bbil  = (const float*)d_in[16];
    float* out = (float*)d_out;

    char* ws = (char*)d_ws;
    size_t off = 0;
    auto alloc = [&](size_t elems, size_t esz) -> char* {
        char* p = ws + off;
        off += (elems * esz + 255) & ~(size_t)255;
        return p;
    };
    // fp32 scratch
    float* e_emb  = (float*)alloc((size_t)N_ * E_ * M_ * D_, 4);
    float* e_att  = (float*)alloc((size_t)N_ * E_ * M_ * (C_ + 1), 4);
    float* glob   = (float*)alloc((size_t)N_ * E_ * D_, 4);
    float* p_hatt = (float*)alloc((size_t)N_ * P_ * M_, 4);
    float* p_tatt = (float*)alloc((size_t)N_ * P_ * M_, 4);
    float* new_hs = (float*)alloc((size_t)N_ * P_ * D_, 4);
    float* new_ts = (float*)alloc((size_t)N_ * P_ * D_, 4);
    float* pa_tmp = (float*)alloc((size_t)N_ * P_ * C_, 4);
    float* rs     = (float*)alloc((size_t)N_ * P_ * D_, 4);
    float* Amat   = (float*)alloc((size_t)N_ * E_ * D_, 4);
    float* Bmat   = (float*)alloc((size_t)N_ * E_ * D_, 4);
    float* edge   = (float*)alloc((size_t)N_ * E_ * E_ * D_, 4);
    float* qW     = (float*)alloc((size_t)N_ * P_ * 4 * D_, 4);
    float* score  = (float*)alloc((size_t)N_ * P_ * E_, 4);
    float* aw     = (float*)alloc((size_t)N_ * P_ * E_, 4);
    float* path2  = (float*)alloc((size_t)N_ * P_ * D_, 4);
    float* hsf    = (float*)alloc((size_t)N_ * P_ * EMB_, 4);
    float* tsf    = (float*)alloc((size_t)N_ * P_ * EMB_, 4);
    // f16 scratch (A operands row-major, B operands transposed N x K)
    h16* glob_h  = (h16*)alloc((size_t)N_ * E_ * D_, 2);
    h16* q_h     = (h16*)alloc((size_t)N_ * P_ * 2 * D_, 2);
    h16* pa_h    = (h16*)alloc((size_t)N_ * P_ * C_, 2);
    h16* seq_t   = (h16*)alloc((size_t)N_ * C_ * D_, 2);         // per doc: D x C
    h16* Wm1_t   = (h16*)alloc((size_t)D_ * D_, 2);
    h16* Wm2_t   = (h16*)alloc((size_t)D_ * D_, 2);
    h16* Watt_t  = (h16*)alloc((size_t)2 * D_ * 4 * D_, 2);      // 3072 x 1536
    h16* Wpath_t = (h16*)alloc((size_t)4 * D_ * D_, 2);          // 768 x 3072
    h16* Whead_t = (h16*)alloc((size_t)3 * D_ * EMB_, 2);        // 768 x 2304
    h16* Wtail_t = (h16*)alloc((size_t)3 * D_ * EMB_, 2);
    h16* Wbil_t  = (h16*)alloc((size_t)EMB_ * BLK_ * L_, 2);     // 97 x 49152
    h16* path_h  = (h16*)alloc((size_t)N_ * P_ * 4 * D_, 2);
    h16* cat_h   = (h16*)alloc((size_t)N_ * P_ * 3 * D_, 2);
    h16* cat_t   = (h16*)alloc((size_t)N_ * P_ * 3 * D_, 2);

    auto tconv = [&](const float* s, h16* d, int K, int Nn, int batch) {
        k_f2h_t<<<dim3((unsigned)((K * Nn + 255) / 256), 1, batch), dim3(256), 0, stream>>>(s, d, K, Nn);
    };
    auto gemm = [&](const h16* A, const h16* Bt, const float* bias, float* C,
                    int Mr, int Nc, int K, int lda, int ldc,
                    long long sA, long long sB, long long sC, int relu, int batch) {
        dim3 g((Nc + 31) / 32, (Mr + 31) / 32, batch);
        k_gemm<<<g, dim3(32), 0, stream>>>(A, Bt, bias, C, Mr, Nc, K, lda, ldc, sA, sB, sC, relu);
    };

    // transpose + f16 conversion of all B operands (one-time, L2 resident afterwards)
    tconv(seq,   seq_t,   C_, D_, N_);
    tconv(Wm1,   Wm1_t,   D_, D_, 1);
    tconv(Wm2,   Wm2_t,   D_, D_, 1);
    tconv(Watt,  Watt_t,  2 * D_, 4 * D_, 1);
    tconv(Wpath, Wpath_t, 4 * D_, D_, 1);
    tconv(Whead, Whead_t, 3 * D_, EMB_, 1);
    tconv(Wtail, Wtail_t, 3 * D_, EMB_, 1);
    tconv(Wbil,  Wbil_t,  EMB_ * BLK_, L_, 1);

    // stage 1: gathers + per-pair attention weights
    k_gather<<<dim3(N_ * E_ * M_), dim3(256), 0, stream>>>(seq, att, ms, e_emb, e_att);
    k_glob<<<dim3((N_ * E_ * D_ + 255) / 256), dim3(256), 0, stream>>>(e_emb, glob, glob_h);
    k_pairw<<<dim3((N_ * P_ + 255) / 256), dim3(256), 0, stream>>>(e_att, ms, hts, p_hatt, p_tatt);
    k_newht<<<dim3(N_ * P_), dim3(256), 0, stream>>>(e_emb, p_hatt, p_tatt, hts, new_hs, new_ts, q_h);
    k_pa<<<dim3(N_ * P_), dim3(256), 0, stream>>>(e_att, p_hatt, p_tatt, hts, pa_tmp, pa_h);

    // stage 2: WMMA GEMMs for rs, glob projections, qW
    gemm(pa_h, seq_t, nullptr, rs, P_, D_, C_, C_, D_,
         (long long)P_ * C_, (long long)C_ * D_, (long long)P_ * D_, 0, N_);
    gemm(glob_h, Wm1_t, nullptr, Amat, E_, D_, D_, D_, D_,
         (long long)E_ * D_, 0, (long long)E_ * D_, 0, N_);
    gemm(glob_h, Wm2_t, nullptr, Bmat, E_, D_, D_, D_, D_,
         (long long)E_ * D_, 0, (long long)E_ * D_, 0, N_);
    k_edge<<<dim3((N_ * E_ * E_ * D_ + 255) / 256), dim3(256), 0, stream>>>(Amat, Bmat, bm, edge);
    gemm(q_h, Watt_t, nullptr, qW, P_, 4 * D_, 2 * D_, 2 * D_, 4 * D_,
         (long long)P_ * 2 * D_, 0, (long long)P_ * 4 * D_, 0, N_);

    // stage 3: score / softmax / path
    k_score<<<dim3((N_ * P_ * E_ + 255) / 256), dim3(256), 0, stream>>>(qW, edge, hts, batt, score);
    k_soft<<<dim3((N_ * P_ + 255) / 256), dim3(256), 0, stream>>>(score, hts, aw);
    k_path<<<dim3(N_ * P_), dim3(256), 0, stream>>>(aw, edge, hts, path_h);

    // stage 4: MLP head GEMMs over all 1520 rows
    gemm(path_h, Wpath_t, bpath, path2, N_ * P_, D_, 4 * D_, 4 * D_, D_, 0, 0, 0, 1, 1);
    k_cat<<<dim3((N_ * P_ * D_ + 255) / 256), dim3(256), 0, stream>>>(new_hs, new_ts, rs, path2, cat_h, cat_t);
    gemm(cat_h, Whead_t, bhead, hsf, N_ * P_, EMB_, 3 * D_, 3 * D_, EMB_, 0, 0, 0, 1, 1);
    gemm(cat_t, Wtail_t, btail, tsf, N_ * P_, EMB_, 3 * D_, 3 * D_, EMB_, 0, 0, 0, 1, 1);

    // stage 5: fused bilinear classifier with async-LDS double buffering
    k_gemm_bil<<<dim3((L_ + 31) / 32, (N_ * P_ + 31) / 32, 1), dim3(32), 0, stream>>>(
        hsf, tsf, Wbil_t, bbil, out, N_ * P_, L_);

    (void)in_sizes; (void)n_in; (void)out_size; (void)ws_size;
}
